// SpatioTemporelLSTM_11458972745752
// MI455X (gfx1250) — matrix operand order repaired and checked by reading, not compiled, not run on hardware
//
#include <hip/hip_runtime.h>
#include <hip/hip_bf16.h>
#include <stdint.h>

// Problem constants (match reference)
#define Bsz  4096
#define Ssz  15
#define Hsz  1024
#define PQn  1003
#define CFGn 128
#define INW  1024        // PQ + 10 + 5 + 3 + 3
#define EPSV 1e-5f

typedef __attribute__((ext_vector_type(16))) __bf16          v16bf;
typedef __attribute__((ext_vector_type(8)))  float           v8f;
typedef __attribute__((ext_vector_type(8)))  unsigned short  us8;

union FragU { us8 u[2]; v16bf v; };

__device__ __forceinline__ float bf2f(unsigned short u) {
    return __uint_as_float(((unsigned)u) << 16);
}
__device__ __forceinline__ unsigned short f2bf(float f) {
    unsigned u = __float_as_uint(f);
    u += 0x7fffu + ((u >> 16) & 1u);           // round-to-nearest-even
    return (unsigned short)(u >> 16);
}
__device__ __forceinline__ float sigm(float x) { return 1.0f / (1.0f + __expf(-x)); }

__device__ __forceinline__ float blockReduceSum(float v, float* sdata) {
    int tid = threadIdx.x;
    sdata[tid] = v; __syncthreads();
    for (int s = 128; s > 0; s >>= 1) {
        if (tid < s) sdata[tid] += sdata[tid + s];
        __syncthreads();
    }
    float r = sdata[0]; __syncthreads();
    return r;
}

// ---------------------------------------------------------------------------
// WMMA bf16 GEMM:  out[M=4096, N] = A[M, K] * Bw^T  (Bw row-major N x K)
// A split into up to 3 segments of width 1024 (row stride 1024), selected by
// k/1024, so concatenated operands ([h|c|x], [xs|h]) need no materialization.
//
// Block = 256 thr = 8 waves (2 in M x 4 in N); block tile 128x128.
// Per 32-deep K stage the block stages A(128x32) and B(128x32) bf16 tiles in
// LDS (row stride padded to 40 elems -> conflict-free ds_load_b128 fragment
// reads), register-double-buffered: global loads for stage s+1 overlap the 8
// v_wmma_f32_16x16x32_bf16 per wave of stage s. One barrier per stage.
// mode 0: outF  = acc + bias                        (gates)
// mode 1: outBf = f2bf(tanh(acc+bias) * x)          (attention score * x)
// ---------------------------------------------------------------------------
#define LDS_STRIDE 40   // 32 + 8 pad (80B rows: bank advance 20 -> no conflicts)

__global__ __launch_bounds__(256)
void gemm_wmma_bf16(const unsigned short* __restrict__ A0,
                    const unsigned short* __restrict__ A1,
                    const unsigned short* __restrict__ A2,
                    const unsigned short* __restrict__ Bw,
                    int K, int N,
                    const float* __restrict__ bias,
                    float* __restrict__ outF,
                    unsigned short* __restrict__ outBf,
                    const unsigned short* __restrict__ xmul,
                    int mode)
{
    __shared__ unsigned short ldsA[2][128 * LDS_STRIDE];   // 2 x 10 KB
    __shared__ unsigned short ldsB[2][128 * LDS_STRIDE];   // 2 x 10 KB

    const int tid   = threadIdx.x;
    const int lane  = tid & 31;
    const int wave  = tid >> 5;
    const int wm    = wave >> 2;                 // 0..1
    const int wn    = wave & 3;                  // 0..3
    const int half  = lane >> 4;                 // 0/1
    const int l16   = lane & 15;
    const int mBlock = blockIdx.y * 128;
    const int nBlock = blockIdx.x * 128;

    // staging map: each thread copies one 32B run of A and one of B per stage
    const int srow = tid >> 1;                   // 0..127
    const int skp  = (tid & 1) * 16;             // 0 or 16 (elements)

    auto loadStage = [&](int kk, us8* ra, us8* rb) {
        const int seg = kk >> 10;
        const unsigned short* As = (seg == 0) ? A0 : ((seg == 1) ? A1 : A2);
        const unsigned short* pa = As + (size_t)(mBlock + srow) * 1024 + (kk & 1023) + skp;
        ra[0] = *(const us8*)pa;
        ra[1] = *(const us8*)(pa + 8);
        const unsigned short* pb = Bw + (size_t)(nBlock + srow) * K + kk + skp;
        rb[0] = *(const us8*)pb;
        rb[1] = *(const us8*)(pb + 8);
    };
    auto storeStage = [&](int buf, const us8* ra, const us8* rb) {
        us8* da = (us8*)&ldsA[buf][srow * LDS_STRIDE + skp];
        da[0] = ra[0]; da[1] = ra[1];
        us8* db = (us8*)&ldsB[buf][srow * LDS_STRIDE + skp];
        db[0] = rb[0]; db[1] = rb[1];
    };

    const v8f vzero = {0.f,0.f,0.f,0.f,0.f,0.f,0.f,0.f};
    v8f acc[4][2];
#pragma unroll
    for (int f = 0; f < 4; ++f)
#pragma unroll
        for (int g = 0; g < 2; ++g) acc[f][g] = vzero;

    us8 ra[2], rb[2];
    loadStage(0, ra, rb);
    storeStage(0, ra, rb);

    const int nStages = K >> 5;
    for (int s = 0; s < nStages; ++s) {
        __syncthreads();                          // buf[s&1] ready for all waves
        const int cur = s & 1;
        const bool more = (s + 1) < nStages;
        if (more) loadStage((s + 1) << 5, ra, rb);

        // A frags: 16x32 bf16; lanes 0-15 hold K[0..7],[16..23]; 16-31 K[8..15],[24..31]
        FragU a[4];
#pragma unroll
        for (int f = 0; f < 4; ++f) {
            const unsigned short* p = &ldsA[cur][(wm * 64 + f * 16 + l16) * LDS_STRIDE];
            a[f].u[0] = *(const us8*)(p + half * 8);
            a[f].u[1] = *(const us8*)(p + 16 + half * 8);
        }
        // B frags: 32x16 bf16; lane holds column n = l16; lanes 0-15 K[0..7],[16..23]; 16-31 K[8..15],[24..31]
        FragU bfr[2];
#pragma unroll
        for (int g = 0; g < 2; ++g) {
            const unsigned short* p = &ldsB[cur][(wn * 32 + g * 16 + l16) * LDS_STRIDE];
            bfr[g].u[0] = *(const us8*)(p + half * 8);
            bfr[g].u[1] = *(const us8*)(p + 16 + half * 8);
        }
#pragma unroll
        for (int f = 0; f < 4; ++f)
#pragma unroll
            for (int g = 0; g < 2; ++g)
                acc[f][g] = __builtin_amdgcn_wmma_f32_16x16x32_bf16(
                    false, a[f].v, false, bfr[g].v, (short)0, acc[f][g], false, false);

        if (more) storeStage((s + 1) & 1, ra, rb);   // other buffer: no barrier needed
    }

    // C/D layout: VGPR r -> M = r (lanes 0-15) / r+8 (lanes 16-31); N = l16
#pragma unroll
    for (int f = 0; f < 4; ++f) {
#pragma unroll
        for (int g = 0; g < 2; ++g) {
            const int col = nBlock + wn * 32 + g * 16 + l16;
            const float bv = bias ? bias[col] : 0.0f;
#pragma unroll
            for (int r = 0; r < 8; ++r) {
                const int row = mBlock + wm * 64 + f * 16 + half * 8 + r;
                const float v = acc[f][g][r] + bv;
                if (mode == 0) {
                    outF[(size_t)row * N + col] = v;
                } else {
                    const float sc = tanhf(v);
                    const float xv = bf2f(xmul[(size_t)row * 1024 + col]);
                    outBf[(size_t)row * 1024 + col] = f2bf(sc * xv);
                }
            }
        }
    }
}

// ---------------------------------------------------------------------------
// Build fused input tensor (S, B, 1024) in bf16:
//   [p_q(1003) | cfg(10)=x_cfg@W_cfg^T+b | emb_hour(5) | emb_wd(3) | emb_sn(3)]
// ---------------------------------------------------------------------------
__global__ void build_inputs(const float* __restrict__ pq, const float* __restrict__ cfgin,
                             const int* __restrict__ timein,
                             const float* __restrict__ Wcfg, const float* __restrict__ bcfg,
                             const float* __restrict__ eh, const float* __restrict__ ew,
                             const float* __restrict__ es,
                             unsigned short* __restrict__ out)
{
    const int bs = blockIdx.x;               // b*S + s
    const int b = bs / Ssz, s = bs % Ssz;
    const int tid = threadIdx.x;             // 128 threads
    const size_t obase = ((size_t)s * Bsz + b) * INW;
    const size_t ibase = (size_t)b * Ssz + s;

    const float* pqr = pq + ibase * PQn;
    for (int i = tid; i < PQn; i += 128) out[obase + i] = f2bf(pqr[i]);

    if (tid < 10) {
        const float* cr = cfgin + ibase * CFGn;
        const float* wr = Wcfg + tid * CFGn;
        float acc = bcfg[tid];
        for (int k = 0; k < CFGn; ++k) acc += cr[k] * wr[k];
        out[obase + PQn + tid] = f2bf(acc);
    }
    const int* tt = timein + ibase * 3;
    if (tid < 5)                 out[obase + 1013 + tid]       = f2bf(eh[tt[0] * 5 + tid]);
    if (tid >= 5 && tid < 8)     out[obase + 1018 + (tid - 5)] = f2bf(ew[tt[1] * 3 + (tid - 5)]);
    if (tid >= 8 && tid < 11)    out[obase + 1021 + (tid - 8)] = f2bf(es[tt[2] * 3 + (tid - 8)]);
}

// --------------------------- small utility kernels -------------------------
__global__ void conv_f2bf(const float* __restrict__ s, unsigned short* __restrict__ d, int n) {
    int i = blockIdx.x * 256 + threadIdx.x;
    if (i < n) d[i] = f2bf(s[i]);
}
__global__ void build_gateW(const float* __restrict__ Wih, const float* __restrict__ Whh,
                            unsigned short* __restrict__ d, int n) {   // n = 4096*2048
    int i = blockIdx.x * 256 + threadIdx.x;
    if (i < n) {
        int row = i >> 11, k = i & 2047;
        float v = (k < 1024) ? Wih[(size_t)row * 1024 + k] : Whh[(size_t)row * 1024 + (k - 1024)];
        d[i] = f2bf(v);
    }
}
__global__ void build_bias(const float* a, const float* b, float* d, int n) {
    int i = blockIdx.x * 256 + threadIdx.x;
    if (i < n) d[i] = a[i] + b[i];
}
__global__ void zero_f32(float* p, int n) {
    int i = blockIdx.x * 256 + threadIdx.x;
    if (i < n) p[i] = 0.0f;
}
__global__ void zero_u16(unsigned short* p, int n) {
    int i = blockIdx.x * 256 + threadIdx.x;
    if (i < n) p[i] = 0;
}

// ------------ step1 elementwise: LSTM cell + mid(bf16) + BN partials -------
__global__ __launch_bounds__(256)
void lstm_ew1(const float* __restrict__ gates, float* __restrict__ cF,
              unsigned short* __restrict__ hB, unsigned short* __restrict__ cB,
              unsigned short* __restrict__ mid, int t,
              float* __restrict__ bnSum, float* __restrict__ bnSqr)
{
    __shared__ float sdata[256];
    const int idx = blockIdx.x * 256 + threadIdx.x;     // b*1024 + h
    const int b = idx >> 10, hh = idx & 1023;
    const float* gr = gates + (size_t)b * 4096;
    const float gi = gr[hh], gf = gr[1024 + hh], gg = gr[2048 + hh], go = gr[3072 + hh];
    const float c = cF[idx];
    const float c2 = sigm(gf) * c + sigm(gi) * tanhf(gg);
    const float h2 = sigm(go) * tanhf(c2);
    cF[idx] = c2; hB[idx] = f2bf(h2); cB[idx] = f2bf(c2);
    mid[(size_t)t * Bsz * Hsz + idx] = f2bf(h2);
    const float s1 = blockReduceSum(h2, sdata);
    const float s2 = blockReduceSum(h2 * h2, sdata);
    if (threadIdx.x == 0) { atomicAdd(&bnSum[t], s1); atomicAdd(&bnSqr[t], s2); }
}

// ---------------- per-timestep batchnorm over (B,H), in place --------------
__global__ void bn_norm(unsigned short* __restrict__ mid,
                        const float* __restrict__ bnSum, const float* __restrict__ bnSqr,
                        const float* __restrict__ gamma, const float* __restrict__ beta)
{
    const int blocksPerS = (Bsz * Hsz) / 256;           // 16384
    const int s = blockIdx.x / blocksPerS;
    const size_t idx = (size_t)blockIdx.x * 256 + threadIdx.x;
    const float invN = 1.0f / ((float)Bsz * (float)Hsz);
    const float mu = bnSum[s] * invN;
    const float var = bnSqr[s] * invN - mu * mu;
    float v = bf2f(mid[idx]);
    v = (v - mu) * rsqrtf(var + EPSV) * gamma[s] + beta[s];
    mid[idx] = f2bf(v);
}

// -- step2 elementwise: LSTM cell + attention partial dot + G=outs@Wih_d^T --
__global__ __launch_bounds__(256)
void lstm_ew2(const float* __restrict__ gates, float* __restrict__ cF,
              unsigned short* __restrict__ hB, unsigned short* __restrict__ cB,
              const unsigned short* __restrict__ xin,    // midn + t*B*H
              const float* __restrict__ Wta,             // (3072)
              const float* __restrict__ Wihd,            // (4,1024)
              float* __restrict__ attAcc, float* __restrict__ gAcc)
{
    __shared__ float sdata[256];
    const int idx = blockIdx.x * 256 + threadIdx.x;
    const int b = idx >> 10, hh = idx & 1023;            // block never straddles b
    const float* gr = gates + (size_t)b * 4096;
    const float gi = gr[hh], gf = gr[1024 + hh], gg = gr[2048 + hh], go = gr[3072 + hh];
    const float c = cF[idx];
    const float c2 = sigm(gf) * c + sigm(gi) * tanhf(gg);
    const float h2 = sigm(go) * tanhf(c2);
    cF[idx] = c2; hB[idx] = f2bf(h2); cB[idx] = f2bf(c2);

    const float xv = bf2f(xin[idx]);
    float r = blockReduceSum(Wta[hh] * h2 + Wta[1024 + hh] * c2 + Wta[2048 + hh] * xv, sdata);
    if (threadIdx.x == 0) atomicAdd(&attAcc[b], r);
#pragma unroll
    for (int g = 0; g < 4; ++g) {
        r = blockReduceSum(Wihd[g * 1024 + hh] * h2, sdata);
        if (threadIdx.x == 0) atomicAdd(&gAcc[b * 4 + g], r);
    }
}

__global__ void att_fin(const float* __restrict__ attAcc, const float* __restrict__ gAcc,
                        const float* __restrict__ bta,
                        float* __restrict__ att2, float* __restrict__ G, int t)
{
    const int b = blockIdx.x * 256 + threadIdx.x;
    if (b < Bsz) {
        att2[(size_t)b * Ssz + t] = tanhf(attAcc[b] + bta[0]);
#pragma unroll
        for (int g = 0; g < 4; ++g)
            G[((size_t)b * Ssz + t) * 4 + g] = gAcc[b * 4 + g];
    }
}

// ------- step3: 15 independent scalar LSTM chains over b = 0..4095 ---------
__global__ void chain_d(const float* __restrict__ G, const float* __restrict__ Whhd,
                        const float* __restrict__ bihd, const float* __restrict__ bhhd,
                        float* __restrict__ out1d)
{
    const int s = threadIdx.x;
    if (s >= Ssz) return;
    float h = 0.f, c = 0.f;
    const float b0 = bihd[0] + bhhd[0], b1 = bihd[1] + bhhd[1];
    const float b2 = bihd[2] + bhhd[2], b3 = bihd[3] + bhhd[3];
    const float w0 = Whhd[0], w1 = Whhd[1], w2 = Whhd[2], w3 = Whhd[3];
    for (int b = 0; b < Bsz; ++b) {
        const float* g4 = G + ((size_t)b * Ssz + s) * 4;
        const float gi = g4[0] + w0 * h + b0;
        const float gf = g4[1] + w1 * h + b1;
        const float gg = g4[2] + w2 * h + b2;
        const float go = g4[3] + w3 * h + b3;
        c = sigm(gf) * c + sigm(gi) * tanhf(gg);
        h = sigm(go) * tanhf(c);
        out1d[(size_t)b * Ssz + s] = h;
    }
}

// ---------------- softmax over S + weighted sum -> out[b] ------------------
__global__ void softmax_dot(const float* __restrict__ att2, const float* __restrict__ out1d,
                            float* __restrict__ out)
{
    const int b = blockIdx.x * 256 + threadIdx.x;
    if (b >= Bsz) return;
    const float* a = att2 + (size_t)b * Ssz;
    const float* o = out1d + (size_t)b * Ssz;
    float m = a[0];
    for (int s = 1; s < Ssz; ++s) m = fmaxf(m, a[s]);
    float sum = 0.f, dot = 0.f;
    for (int s = 0; s < Ssz; ++s) {
        const float e = __expf(a[s] - m);
        sum += e; dot += e * o[s];
    }
    out[b] = dot / sum;
}

// ===========================================================================
extern "C" void kernel_launch(void* const* d_in, const int* in_sizes, int n_in,
                              void* d_out, int out_size, void* d_ws, size_t ws_size,
                              hipStream_t stream)
{
    (void)in_sizes; (void)n_in; (void)out_size; (void)ws_size;
    const float* in_pq   = (const float*)d_in[0];
    const float* in_cfg  = (const float*)d_in[1];
    const int*   in_time = (const int*)  d_in[2];
    const float* W_cfg   = (const float*)d_in[3];
    const float* b_cfg   = (const float*)d_in[4];
    const float* emb_h   = (const float*)d_in[5];
    const float* emb_w   = (const float*)d_in[6];
    const float* emb_s   = (const float*)d_in[7];
    const float* W_sa    = (const float*)d_in[8];
    const float* b_sa    = (const float*)d_in[9];
    const float* Wih_s   = (const float*)d_in[10];
    const float* Whh_s   = (const float*)d_in[11];
    const float* bih_s   = (const float*)d_in[12];
    const float* bhh_s   = (const float*)d_in[13];
    const float* Wih_t   = (const float*)d_in[14];
    const float* Whh_t   = (const float*)d_in[15];
    const float* bih_t   = (const float*)d_in[16];
    const float* bhh_t   = (const float*)d_in[17];
    const float* W_ta    = (const float*)d_in[18];
    const float* b_ta    = (const float*)d_in[19];
    const float* bn_g    = (const float*)d_in[20];
    const float* bn_b    = (const float*)d_in[21];
    const float* Wih_d   = (const float*)d_in[22];
    const float* Whh_d   = (const float*)d_in[23];
    const float* bih_d   = (const float*)d_in[24];
    const float* bhh_d   = (const float*)d_in[25];

    // ---- workspace carve-up (~402 MB total) ----
    char* ws = (char*)d_ws; size_t off = 0;
    auto alloc = [&](size_t bytes) -> char* {
        char* p = ws + off; off += (bytes + 255) & ~(size_t)255; return p;
    };
    unsigned short* inBf  = (unsigned short*)alloc((size_t)Ssz * Bsz * INW * 2);   // (S,B,1024)
    unsigned short* midBf = (unsigned short*)alloc((size_t)Ssz * Bsz * Hsz * 2);   // (S,B,1024)
    unsigned short* WsaBf = (unsigned short*)alloc((size_t)1024 * 3072 * 2);
    unsigned short* WgS   = (unsigned short*)alloc((size_t)4096 * 2048 * 2);
    unsigned short* WgT   = (unsigned short*)alloc((size_t)4096 * 2048 * 2);
    float*          biasS = (float*)alloc(4096 * 4);
    float*          biasT = (float*)alloc(4096 * 4);
    float*          cF    = (float*)alloc((size_t)Bsz * Hsz * 4);
    unsigned short* hB    = (unsigned short*)alloc((size_t)2 * Bsz * Hsz * 2);     // hB|cB contiguous
    unsigned short* cB    = hB + (size_t)Bsz * Hsz;
    unsigned short* xsBf  = (unsigned short*)alloc((size_t)Bsz * Hsz * 2);
    float*          gates = (float*)alloc((size_t)Bsz * 4096 * 4);
    float*          bnSum = (float*)alloc(32 * 4);                                  // [0..14]=sum
    float*          bnSqr = bnSum + 16;                                             // [16..30]=sumsq
    float*          attAcc= (float*)alloc((size_t)Bsz * 5 * 4);                     // attAcc|gAcc
    float*          gAcc  = attAcc + Bsz;
    float*          att2  = (float*)alloc((size_t)Bsz * Ssz * 4);
    float*          Gbuf  = (float*)alloc((size_t)Bsz * Ssz * 4 * 4);
    float*          out1d = (float*)alloc((size_t)Bsz * Ssz * 4);

    const int T256 = 256;
    // ---- prep: fused inputs + bf16 weights + combined biases ----
    build_inputs<<<Bsz * Ssz, 128, 0, stream>>>(in_pq, in_cfg, in_time, W_cfg, b_cfg,
                                                emb_h, emb_w, emb_s, inBf);
    conv_f2bf<<<(1024 * 3072 + 255) / 256, T256, 0, stream>>>(W_sa, WsaBf, 1024 * 3072);
    build_gateW<<<(4096 * 2048 + 255) / 256, T256, 0, stream>>>(Wih_s, Whh_s, WgS, 4096 * 2048);
    build_gateW<<<(4096 * 2048 + 255) / 256, T256, 0, stream>>>(Wih_t, Whh_t, WgT, 4096 * 2048);
    build_bias<<<16, T256, 0, stream>>>(bih_s, bhh_s, biasS, 4096);
    build_bias<<<16, T256, 0, stream>>>(bih_t, bhh_t, biasT, 4096);
    zero_f32<<<(Bsz * Hsz + 255) / 256, T256, 0, stream>>>(cF, Bsz * Hsz);
    zero_u16<<<(2 * Bsz * Hsz + 255) / 256, T256, 0, stream>>>(hB, 2 * Bsz * Hsz);
    zero_f32<<<1, 32, 0, stream>>>(bnSum, 32);

    const dim3 gScore(1024 / 128, Bsz / 128);   // (8, 32)
    const dim3 gGates(4096 / 128, Bsz / 128);   // (32, 32)
    const int ewBlocks = (Bsz * Hsz) / 256;

    // ---- stage 1: spatial-attention LSTM over S=15 ----
    for (int t = 0; t < Ssz; ++t) {
        const unsigned short* xt = inBf + (size_t)t * Bsz * INW;
        // score = tanh([h|c|x] @ W_sa^T + b_sa); xs = x * score (bf16)
        gemm_wmma_bf16<<<gScore, T256, 0, stream>>>(hB, cB, xt, WsaBf, 3072, 1024,
                                                    b_sa, nullptr, xsBf, xt, 1);
        // gates = [xs|h] @ [Wih_s|Whh_s]^T + (bih+bhh)
        gemm_wmma_bf16<<<gGates, T256, 0, stream>>>(xsBf, hB, nullptr, WgS, 2048, 4096,
                                                    biasS, gates, nullptr, nullptr, 0);
        lstm_ew1<<<ewBlocks, T256, 0, stream>>>(gates, cF, hB, cB, midBf, t, bnSum, bnSqr);
    }
    // ---- batchnorm over (B,H) per timestep, in place on bf16 mid ----
    bn_norm<<<Ssz * ewBlocks, T256, 0, stream>>>(midBf, bnSum, bnSqr, bn_g, bn_b);

    // ---- stage 2: temporal LSTM + attention scores + G = outs @ Wih_d^T ----
    zero_f32<<<(Bsz * Hsz + 255) / 256, T256, 0, stream>>>(cF, Bsz * Hsz);
    zero_u16<<<(2 * Bsz * Hsz + 255) / 256, T256, 0, stream>>>(hB, 2 * Bsz * Hsz);
    for (int t = 0; t < Ssz; ++t) {
        const unsigned short* xt = midBf + (size_t)t * Bsz * Hsz;
        zero_f32<<<(Bsz * 5 + 255) / 256, T256, 0, stream>>>(attAcc, Bsz * 5);
        gemm_wmma_bf16<<<gGates, T256, 0, stream>>>(xt, hB, nullptr, WgT, 2048, 4096,
                                                    biasT, gates, nullptr, nullptr, 0);
        lstm_ew2<<<ewBlocks, T256, 0, stream>>>(gates, cF, hB, cB, xt, W_ta, Wih_d,
                                                attAcc, gAcc);
        att_fin<<<(Bsz + 255) / 256, T256, 0, stream>>>(attAcc, gAcc, b_ta, att2, Gbuf, t);
    }

    // ---- stage 3: 15 parallel scalar chains over B, then softmax+dot ----
    chain_d<<<1, 32, 0, stream>>>(Gbuf, Whh_d, bih_d, bhh_d, out1d);
    softmax_dot<<<(Bsz + 255) / 256, T256, 0, stream>>>(att2, out1d, (float*)d_out);
}
